// Gudi_UpProj_Block_31696858644918
// MI455X (gfx1250) — compile-verified, matrix operand order checked by reading, not run on hardware
//
#include <hip/hip_runtime.h>

// ---------------- types ----------------
typedef __bf16 v16bf __attribute__((ext_vector_type(16)));
typedef float  v8f   __attribute__((ext_vector_type(8)));

#define DEVINL __device__ __forceinline__

DEVINL unsigned short f2bf(float f) {
    union { float f; unsigned u; } v; v.f = f;
    unsigned r = v.u + 0x7fffu + ((v.u >> 16) & 1u);   // round-to-nearest-even
    return (unsigned short)(r >> 16);
}

// Per-parity-class packed weight base offsets (elements). class = ph*2+pw
DEVINL long wclassBase(int cls) {
    const long K0 = 512L * 9216, K1 = 512L * 6144, K2 = 512L * 6144;
    switch (cls) { case 0: return 0; case 1: return K0; case 2: return K0 + K1; default: return K0 + K1 + K2; }
}
__host__ __device__ constexpr long wclassBaseC(int cls) {
    return cls == 0 ? 0L
         : cls == 1 ? 512L * 9216
         : cls == 2 ? 512L * 9216 + 512L * 6144
         :            512L * 9216 + 512L * 6144 + 512L * 6144;
}

// ---------------- packing kernels ----------------

// x: NCHW fp32 [16,1024,16,16] -> NHWC bf16 [16,16,16,1024]
__global__ __launch_bounds__(256) void pack_x(const float* __restrict__ x, unsigned short* __restrict__ xbf) {
    long idx = (long)blockIdx.x * 256 + threadIdx.x;        // 4,194,304
    int w  = (int)(idx & 15);
    int h  = (int)((idx >> 4) & 15);
    int c  = (int)((idx >> 8) & 1023);
    int b  = (int)(idx >> 18);
    xbf[((((long)b * 16 + h) * 16 + w) << 10) + c] = f2bf(x[idx]);
}

// 5x5 OIHW fp32 -> per-class [Cout][tap*1024+ci] bf16
__global__ __launch_bounds__(256) void pack_w5(const float* __restrict__ w, unsigned short* __restrict__ wp) {
    long idx = (long)blockIdx.x * 256 + threadIdx.x;        // 13,107,200
    int kw = (int)(idx % 5);
    long t = idx / 5;
    int kh = (int)(t % 5);
    t /= 5;
    int ci = (int)(t & 1023);
    int co = (int)(t >> 10);
    int ph = kh & 1, pw = kw & 1;
    int cls = ph * 2 + pw;
    int nH = ph ? 2 : 3, nW = pw ? 2 : 3;
    int tap = (kh >> 1) * nW + (kw >> 1);
    long Kc = 1024L * nH * nW;
    wp[wclassBase(cls) + (long)co * Kc + (long)tap * 1024 + ci] = f2bf(w[idx]);
}

// 3x3 OIHW fp32 -> [Cout][(kh*3+kw)*512+ci] bf16
__global__ __launch_bounds__(256) void pack_w3(const float* __restrict__ w, unsigned short* __restrict__ wp) {
    long idx = (long)blockIdx.x * 256 + threadIdx.x;        // 2,359,296
    int kw = (int)(idx % 3);
    long t = idx / 3;
    int kh = (int)(t % 3);
    t /= 3;
    int ci = (int)(t & 511);
    int co = (int)(t >> 9);
    wp[(long)co * 4608 + (long)(kh * 3 + kw) * 512 + ci] = f2bf(w[idx]);
}

// ---------------- conv1 + shortcut fused implicit GEMM, templated per parity class ----------------
// grid: (Ntiles=32, Mtiles=8), block 256 ; block tile 64M x 128N ; wave tile 32x32, dual output
// Dynamic LDS only (segment base = offset 0) so raw LDS byte offsets feed
// GLOBAL_LOAD_ASYNC_TO_LDS_B128. Layout:
//   [0, 8192)        : A1 tiles, 2 bufs x 64*32*2B   (buf stride 4096)
//   [8192, 16384)    : A2 tiles, 2 bufs x 64*32*2B   (buf stride 4096)
//   [16384, 32768)   : B tiles,  2 bufs x 128*32*2B  (buf stride 8192)
#define CONV1_LDS_BYTES 32768
template <int PH, int PW>
__global__ __launch_bounds__(256) void conv1sc_gemm_t(
    const unsigned short* __restrict__ xbf,
    const unsigned short* __restrict__ w1p,
    const unsigned short* __restrict__ wscp,
    float* __restrict__ raw1, float* __restrict__ rawsc)
{
    constexpr int NW = PW ? 2 : 3;
    constexpr int NH = PH ? 2 : 3;
    constexpr int KC = 1024 * NH * NW;
    constexpr long BASE = wclassBaseC(PH * 2 + PW);
    constexpr unsigned LA_STRIDE = 4096;
    constexpr unsigned LA2_BASE  = 8192;
    constexpr unsigned LB_BASE   = 16384;
    constexpr unsigned LB_STRIDE = 8192;

    extern __shared__ __attribute__((aligned(128))) char smem[];
    unsigned short* sA1 = (unsigned short*)smem;                 // elem idx: buf*2048 + off
    unsigned short* sA2 = (unsigned short*)(smem + LA2_BASE);    // elem idx: buf*2048 + off
    unsigned short* sB  = (unsigned short*)(smem + LB_BASE);     // elem idx: buf*4096 + off

    const int tid = threadIdx.x, lane = tid & 31, wave = tid >> 5;
    const int waveM = wave >> 2, waveN = wave & 3;
    const int blockM = blockIdx.y * 64, blockN = blockIdx.x * 128;

    // A staging: row 0..63, 8-elem quarter
    const int ar = tid >> 2, aq = tid & 3;
    const unsigned short* __restrict__ pA1 = w1p + BASE + (long)(blockM + ar) * KC + aq * 8;
    const unsigned short* __restrict__ pA2 = wscp + BASE + (long)(blockM + ar) * KC + aq * 8;
    const int aOff = ar * 32 + aq * 8;

    // B staging: 2 slots, per-slot base pointer + (i,j)
    const unsigned short* pB[2];
    int iS[2], jS[2], qS[2], bOff[2];
    #pragma unroll
    for (int s = 0; s < 2; ++s) {
        int slot = tid + s * 256;
        int row = slot >> 2, q = slot & 3;
        int n = blockN + row;
        int b = n >> 8, i = (n >> 4) & 15, j = n & 15;
        pB[s] = xbf + ((((long)b * 16 + i) * 16 + j) << 10);
        iS[s] = i; jS[s] = j; qS[s] = q;
        bOff[s] = row * 32 + q * 8;
    }

    int aFr[2], bFr[2];
    #pragma unroll
    for (int mi = 0; mi < 2; ++mi) aFr[mi] = (waveM * 32 + mi * 16 + (lane & 15)) * 32 + (lane >> 4) * 16;
    #pragma unroll
    for (int ni = 0; ni < 2; ++ni) bFr[ni] = (waveN * 32 + ni * 16 + (lane & 15)) * 32 + (lane >> 4) * 16;

    v8f acc1[4], acc2[4];
    #pragma unroll
    for (int i = 0; i < 4; ++i)
        #pragma unroll
        for (int j = 0; j < 8; ++j) { acc1[i][j] = 0.f; acc2[i][j] = 0.f; }

    // async staging: weights dense DMA; activations DMA for valid lanes, ds_store zeros for OOB
    auto stage = [&](int buf, int k0) {
        unsigned dstA = (unsigned)buf * LA_STRIDE + (unsigned)(aOff * 2);
        asm volatile("global_load_async_to_lds_b128 %0, %1, off"
                     :: "v"(dstA), "v"((unsigned long long)(const void*)(pA1 + k0)) : "memory");
        asm volatile("global_load_async_to_lds_b128 %0, %1, off"
                     :: "v"(dstA + LA2_BASE), "v"((unsigned long long)(const void*)(pA2 + k0)) : "memory");
        __builtin_prefetch(pA1 + k0 + 256, 0, 1);
        __builtin_prefetch(pA2 + k0 + 256, 0, 1);
        #pragma unroll
        for (int s = 0; s < 2; ++s) {
            int k = k0 + qS[s] * 8;
            int tap = k >> 10, ci = k & 1023;
            int tH, tW;
            if constexpr (NW == 2) { tH = tap >> 1; tW = tap & 1; }
            else                   { tH = tap / 3;  tW = tap - 3 * tH; }   // constant divisor
            int dh = tH - (PH ? 0 : 1);
            int dw = tW - (PW ? 0 : 1);
            int ih = iS[s] + dh, jw = jS[s] + dw;
            unsigned dstB = LB_BASE + (unsigned)buf * LB_STRIDE + (unsigned)(bOff[s] * 2);
            if ((unsigned)ih < 16u && (unsigned)jw < 16u) {
                const unsigned short* src = pB[s] + (((long)(dh * 16 + dw)) << 10) + ci;
                asm volatile("global_load_async_to_lds_b128 %0, %1, off"
                             :: "v"(dstB), "v"((unsigned long long)(const void*)src) : "memory");
            } else {
                *(uint4*)&sB[(unsigned)buf * 4096u + (unsigned)bOff[s]] = make_uint4(0u, 0u, 0u, 0u);
            }
        }
    };
    auto compute = [&](int buf) {
        v16bf af1[2], af2[2], bf[2];
        #pragma unroll
        for (int mi = 0; mi < 2; ++mi) {
            af1[mi] = *(const v16bf*)&sA1[buf * 2048 + aFr[mi]];
            af2[mi] = *(const v16bf*)&sA2[buf * 2048 + aFr[mi]];
        }
        #pragma unroll
        for (int ni = 0; ni < 2; ++ni) bf[ni] = *(const v16bf*)&sB[buf * 4096 + bFr[ni]];
        #pragma unroll
        for (int mi = 0; mi < 2; ++mi)
            #pragma unroll
            for (int ni = 0; ni < 2; ++ni) {
                acc1[mi * 2 + ni] = __builtin_amdgcn_wmma_f32_16x16x32_bf16(
                    false, af1[mi], false, bf[ni], (short)0, acc1[mi * 2 + ni], false, false);
                acc2[mi * 2 + ni] = __builtin_amdgcn_wmma_f32_16x16x32_bf16(
                    false, af2[mi], false, bf[ni], (short)0, acc2[mi * 2 + ni], false, false);
            }
    };

    constexpr int KT = KC / 32;
    stage(0, 0);
    asm volatile("s_wait_asynccnt 0" ::: "memory");
    #pragma unroll 1
    for (int kt = 0; kt < KT; ++kt) {
        __syncthreads();                          // publishes buffer `cur`
        int cur = kt & 1;
        bool more = (kt + 1) < KT;
        if (more) stage(cur ^ 1, (kt + 1) * 32);  // DMA next tile while WMMAs run
        compute(cur);
        if (more) asm volatile("s_wait_asynccnt 0" ::: "memory");
    }

    // store: element e of v8f in lane L -> M = e + 8*(L>>4), N = L&15
    #pragma unroll
    for (int mi = 0; mi < 2; ++mi)
        #pragma unroll
        for (int ni = 0; ni < 2; ++ni) {
            int coB = blockM + waveM * 32 + mi * 16 + ((lane >> 4) << 3);
            int n = blockN + waveN * 32 + ni * 16 + (lane & 15);
            int b = n >> 8, i = (n >> 4) & 15, j = n & 15;
            int oh = 2 * i + PH, ow = 2 * j + PW;
            long nfull = ((long)b * 32 + oh) * 32 + ow;
            float* p1 = raw1 + nfull * 512 + coB;
            float* p2 = rawsc + nfull * 512 + coB;
            v8f& A = acc1[mi * 2 + ni];
            v8f& C = acc2[mi * 2 + ni];
            *(float4*)(p1)     = make_float4(A[0], A[1], A[2], A[3]);
            *(float4*)(p1 + 4) = make_float4(A[4], A[5], A[6], A[7]);
            *(float4*)(p2)     = make_float4(C[0], C[1], C[2], C[3]);
            *(float4*)(p2 + 4) = make_float4(C[4], C[5], C[6], C[7]);
        }
}

// ---------------- conv2 implicit GEMM (3x3 dense), wave tile 32x64, async-to-LDS staging ----------------
// grid: (Ntiles=64, Mtiles=8), block 256 ; block tile 64M x 256N
//   [0, 8192)      : A tiles, 2 buffers x 64*32*2B   (buf stride 4096)
//   [8192, 40960)  : B tiles, 2 buffers x 256*32*2B  (buf stride 16384)
#define CONV2_LDS_BYTES 40960
__global__ __launch_bounds__(256) void conv2_gemm(
    const unsigned short* __restrict__ y1bf,
    const unsigned short* __restrict__ w2p,
    float* __restrict__ raw2)
{
    constexpr int KC = 4608;
    constexpr unsigned LA_STRIDE = 4096;     // bytes per A buffer
    constexpr unsigned LB_BASE   = 8192;     // byte base of B region
    constexpr unsigned LB_STRIDE = 16384;    // bytes per B buffer

    extern __shared__ __attribute__((aligned(128))) char smem[];
    unsigned short* sA = (unsigned short*)smem;            // element idx: buf*2048 + off
    unsigned short* sB = (unsigned short*)(smem + LB_BASE);// element idx: buf*8192 + off

    const int tid = threadIdx.x, lane = tid & 31, wave = tid >> 5;
    const int waveM = wave >> 2, waveN = wave & 3;
    const int blockM = blockIdx.y * 64, blockN = blockIdx.x * 256;

    // A staging
    const int ar = tid >> 2, aq = tid & 3;
    const unsigned short* __restrict__ pA = w2p + (long)(blockM + ar) * KC + aq * 8;
    const int aOff = ar * 32 + aq * 8;

    // B staging: 4 slots
    const unsigned short* pB[4];
    int ohS[4], owS[4], qS[4], bOff[4];
    #pragma unroll
    for (int s = 0; s < 4; ++s) {
        int slot = tid + s * 256;
        int row = slot >> 2, q = slot & 3;
        int n = blockN + row;                       // 0..16383
        int b = n >> 10, oh = (n >> 5) & 31, ow = n & 31;
        pB[s] = y1bf + ((((long)b * 32 + oh) * 32 + ow) << 9);
        ohS[s] = oh; owS[s] = ow; qS[s] = q;
        bOff[s] = row * 32 + q * 8;
    }

    int aFr[2], bFr[4];
    #pragma unroll
    for (int mi = 0; mi < 2; ++mi) aFr[mi] = (waveM * 32 + mi * 16 + (lane & 15)) * 32 + (lane >> 4) * 16;
    #pragma unroll
    for (int ni = 0; ni < 4; ++ni) bFr[ni] = (waveN * 64 + ni * 16 + (lane & 15)) * 32 + (lane >> 4) * 16;

    v8f acc[8];
    #pragma unroll
    for (int i = 0; i < 8; ++i)
        #pragma unroll
        for (int j = 0; j < 8; ++j) acc[i][j] = 0.f;

    auto stage = [&](int buf, int k0) {
        unsigned dstA = (unsigned)buf * LA_STRIDE + (unsigned)(aOff * 2);
        asm volatile("global_load_async_to_lds_b128 %0, %1, off"
                     :: "v"(dstA), "v"((unsigned long long)(const void*)(pA + k0)) : "memory");
        #pragma unroll
        for (int s = 0; s < 4; ++s) {
            int k = k0 + qS[s] * 8;
            int tap = k >> 9, ci = k & 511;
            int kh = tap / 3, kw = tap - 3 * kh;    // constant divisor
            int ih = ohS[s] + kh - 1, jw = owS[s] + kw - 1;
            unsigned dstB = LB_BASE + (unsigned)buf * LB_STRIDE + (unsigned)(bOff[s] * 2);
            if ((unsigned)ih < 32u && (unsigned)jw < 32u) {
                const unsigned short* src = pB[s] + (((long)((kh - 1) * 32 + (kw - 1))) << 9) + ci;
                asm volatile("global_load_async_to_lds_b128 %0, %1, off"
                             :: "v"(dstB), "v"((unsigned long long)(const void*)src) : "memory");
            } else {
                *(uint4*)&sB[(unsigned)buf * 8192u + (unsigned)bOff[s]] = make_uint4(0u, 0u, 0u, 0u);
            }
        }
    };
    auto compute = [&](int buf) {
        v16bf af[2], bf[4];
        #pragma unroll
        for (int mi = 0; mi < 2; ++mi) af[mi] = *(const v16bf*)&sA[buf * 2048 + aFr[mi]];
        #pragma unroll
        for (int ni = 0; ni < 4; ++ni) bf[ni] = *(const v16bf*)&sB[buf * 8192 + bFr[ni]];
        #pragma unroll
        for (int mi = 0; mi < 2; ++mi)
            #pragma unroll
            for (int ni = 0; ni < 4; ++ni)
                acc[mi * 4 + ni] = __builtin_amdgcn_wmma_f32_16x16x32_bf16(
                    false, af[mi], false, bf[ni], (short)0, acc[mi * 4 + ni], false, false);
    };

    constexpr int KT = KC / 32;   // 144
    stage(0, 0);
    asm volatile("s_wait_asynccnt 0" ::: "memory");
    #pragma unroll 1
    for (int kt = 0; kt < KT; ++kt) {
        __syncthreads();                       // publishes buffer `cur`
        int cur = kt & 1;
        bool more = (kt + 1) < KT;
        if (more) stage(cur ^ 1, (kt + 1) * 32);  // DMA next tile while WMMAs run
        compute(cur);
        if (more) asm volatile("s_wait_asynccnt 0" ::: "memory");
    }

    #pragma unroll
    for (int mi = 0; mi < 2; ++mi)
        #pragma unroll
        for (int ni = 0; ni < 4; ++ni) {
            int coB = blockM + waveM * 32 + mi * 16 + ((lane >> 4) << 3);
            int n = blockN + waveN * 64 + ni * 16 + (lane & 15);
            float* p = raw2 + (long)n * 512 + coB;
            v8f& A = acc[mi * 4 + ni];
            *(float4*)(p)     = make_float4(A[0], A[1], A[2], A[3]);
            *(float4*)(p + 4) = make_float4(A[4], A[5], A[6], A[7]);
        }
}

// ---------------- BN stats: per-channel mean/var -> scale/bias ----------------
__global__ __launch_bounds__(256) void bnstats(const float* __restrict__ raw,
                                               const float* __restrict__ gamma,
                                               const float* __restrict__ beta,
                                               float* __restrict__ sc, float* __restrict__ bi)
{
    const int co = blockIdx.x;          // 512
    const int tid = threadIdx.x;
    float s = 0.f, s2 = 0.f;
    for (int n = tid; n < 16384; n += 256) {
        float v = raw[(long)n * 512 + co];
        s += v; s2 += v * v;
    }
    __shared__ float sh[256], sh2[256];
    sh[tid] = s; sh2[tid] = s2;
    __syncthreads();
    for (int o = 128; o > 0; o >>= 1) {
        if (tid < o) { sh[tid] += sh[tid + o]; sh2[tid] += sh2[tid + o]; }
        __syncthreads();
    }
    if (tid == 0) {
        float mean = sh[0] * (1.f / 16384.f);
        float var  = sh2[0] * (1.f / 16384.f) - mean * mean;   // biased, matches jnp.var
        float a = gamma[co] * rsqrtf(var + 1e-5f);
        sc[co] = a;
        bi[co] = beta[co] - a * mean;
    }
}

// y1 = bf16(relu(a*raw1 + b))  (NHWC)
__global__ __launch_bounds__(256) void apply_bn_relu(const float* __restrict__ raw1,
                                                     const float* __restrict__ sc,
                                                     const float* __restrict__ bi,
                                                     unsigned short* __restrict__ y1bf)
{
    long idx = (long)blockIdx.x * 256 + threadIdx.x;   // 8,388,608
    int co = (int)(idx & 511);
    float v = sc[co] * raw1[idx] + bi[co];
    y1bf[idx] = f2bf(v > 0.f ? v : 0.f);
}

// out = relu(bn2(raw2) + bnsc(rawsc)) ; NHWC -> NCHW fp32
__global__ __launch_bounds__(256) void final_combine(const float* __restrict__ raw2,
                                                     const float* __restrict__ rawsc,
                                                     const float* __restrict__ a2, const float* __restrict__ b2,
                                                     const float* __restrict__ asc, const float* __restrict__ bsc,
                                                     float* __restrict__ out)
{
    long idx = (long)blockIdx.x * 256 + threadIdx.x;   // 8,388,608
    int co = (int)(idx & 511);
    long n = idx >> 9;
    float v = a2[co] * raw2[idx] + b2[co] + asc[co] * rawsc[idx] + bsc[co];
    v = v > 0.f ? v : 0.f;
    long b = n >> 10, hw = n & 1023;                   // 32*32 = 1024
    out[(((long)b * 512 + co) << 10) + hw] = v;
}

// ---------------- launcher ----------------
extern "C" void kernel_launch(void* const* d_in, const int* in_sizes, int n_in,
                              void* d_out, int out_size, void* d_ws, size_t ws_size,
                              hipStream_t stream)
{
    (void)in_sizes; (void)n_in; (void)out_size; (void)ws_size;
    const float* x   = (const float*)d_in[0];
    const float* w1  = (const float*)d_in[1];
    const float* g1  = (const float*)d_in[2];
    const float* b1  = (const float*)d_in[3];
    const float* w2  = (const float*)d_in[4];
    const float* g2  = (const float*)d_in[5];
    const float* b2  = (const float*)d_in[6];
    const float* wsc = (const float*)d_in[7];
    const float* gsc = (const float*)d_in[8];
    const float* bsc = (const float*)d_in[9];
    float* out = (float*)d_out;

    char* ws = (char*)d_ws;
    size_t off = 0;
    auto alloc = [&](size_t bytes) { size_t r = off; off += (bytes + 255) & ~(size_t)255; return r; };
    unsigned short* xbf  = (unsigned short*)(ws + alloc(4194304ULL * 2));
    unsigned short* w1p  = (unsigned short*)(ws + alloc(13107200ULL * 2));
    unsigned short* wscp = (unsigned short*)(ws + alloc(13107200ULL * 2));
    unsigned short* w2p  = (unsigned short*)(ws + alloc(2359296ULL * 2));
    float* raw1  = (float*)(ws + alloc(8388608ULL * 4));
    float* rawsc = (float*)(ws + alloc(8388608ULL * 4));
    float* raw2  = (float*)(ws + alloc(8388608ULL * 4));
    unsigned short* y1bf = (unsigned short*)(ws + alloc(8388608ULL * 2));
    float* sc1  = (float*)(ws + alloc(512 * 4));
    float* bi1  = (float*)(ws + alloc(512 * 4));
    float* scS  = (float*)(ws + alloc(512 * 4));
    float* biS  = (float*)(ws + alloc(512 * 4));
    float* sc2  = (float*)(ws + alloc(512 * 4));
    float* bi2  = (float*)(ws + alloc(512 * 4));

    // pack to bf16
    pack_x<<<16384, 256, 0, stream>>>(x, xbf);
    pack_w5<<<51200, 256, 0, stream>>>(w1, w1p);
    pack_w5<<<51200, 256, 0, stream>>>(wsc, wscp);
    pack_w3<<<9216, 256, 0, stream>>>(w2, w2p);

    // conv1 + shortcut fused, one specialized kernel per parity class, async-to-LDS staging
    dim3 g1g(32, 8);
    conv1sc_gemm_t<0, 0><<<g1g, 256, CONV1_LDS_BYTES, stream>>>(xbf, w1p, wscp, raw1, rawsc);
    conv1sc_gemm_t<0, 1><<<g1g, 256, CONV1_LDS_BYTES, stream>>>(xbf, w1p, wscp, raw1, rawsc);
    conv1sc_gemm_t<1, 0><<<g1g, 256, CONV1_LDS_BYTES, stream>>>(xbf, w1p, wscp, raw1, rawsc);
    conv1sc_gemm_t<1, 1><<<g1g, 256, CONV1_LDS_BYTES, stream>>>(xbf, w1p, wscp, raw1, rawsc);

    // BN stats + apply
    bnstats<<<512, 256, 0, stream>>>(raw1, g1, b1, sc1, bi1);
    bnstats<<<512, 256, 0, stream>>>(rawsc, gsc, bsc, scS, biS);
    apply_bn_relu<<<32768, 256, 0, stream>>>(raw1, sc1, bi1, y1bf);

    // conv2: async direct-to-LDS staging + wave tile 32x64
    conv2_gemm<<<dim3(64, 8), 256, CONV2_LDS_BYTES, stream>>>(y1bf, w2p, raw2);
    bnstats<<<512, 256, 0, stream>>>(raw2, g2, b2, sc2, bi2);

    // out = relu(bn2 + bnsc), NHWC->NCHW
    final_combine<<<32768, 256, 0, stream>>>(raw2, rawsc, sc2, bi2, scS, biS, out);
}
